// WeightOnlyQLinear_2095944040881
// MI455X (gfx1250) — matrix operand order repaired and checked by reading, not compiled, not run on hardware
//
#include <hip/hip_runtime.h>

typedef __attribute__((ext_vector_type(16))) _Float16      v16h;
typedef __attribute__((ext_vector_type(8)))  float         v8f;
typedef __attribute__((ext_vector_type(8)))  unsigned short u16x8;

struct U16x16 { u16x8 lo, hi; };

__device__ __forceinline__ unsigned short f2h(float f) {
    _Float16 h = (_Float16)f;               // RNE convert
    return __builtin_bit_cast(unsigned short, h);
}
__device__ __forceinline__ unsigned pk2(float lo, float hi) {
    return (unsigned)f2h(lo) | ((unsigned)f2h(hi) << 16);
}

#define BM 128
#define BN 128
#define BK 64
#define ASTR 72   // padded LDS stride (halves): 144B row stride -> conflict-free frag loads
#define BSTR 72

__global__ __launch_bounds__(256)
void w4a16_wmma_gemm(const float* __restrict__ x,
                     const int*   __restrict__ qw,
                     const int*   __restrict__ qz,
                     const float* __restrict__ sc,
                     const float* __restrict__ bias,
                     float* __restrict__ out,
                     int M, int N, int K, int NP8, int gs)
{
    __shared__ __align__(16) unsigned short As[BM * ASTR]; // x tile, row-major [m][k]
    __shared__ __align__(16) unsigned short Bs[BN * BSTR]; // w tile, col-major [n][k]

    const int tid  = threadIdx.x;
    const int lane = tid & 31;
    const int wave = tid >> 5;
    const int wm   = wave >> 2;      // 0..1 : which 64-row slab
    const int wn   = wave & 3;       // 0..3 : which 32-col slab
    const int m0   = blockIdx.y * BM;
    const int n0   = blockIdx.x * BN;

    // dequant ownership (constant per thread): one int32 column, 4 k-rows
    const int ng  = tid & 15;        // packed-int32 column within the 128-wide tile
    const int kr0 = tid >> 4;        // base k-row (0..15), step 16

    v8f acc[4][2];
    #pragma unroll
    for (int mt = 0; mt < 4; ++mt)
        #pragma unroll
        for (int nt = 0; nt < 2; ++nt)
            acc[mt][nt] = (v8f){0.f,0.f,0.f,0.f,0.f,0.f,0.f,0.f};

    for (int kt = 0; kt < K; kt += BK) {
        // ---- stage A: x (f32) -> f16 tile in LDS ----
        #pragma unroll
        for (int i = 0; i < 8; ++i) {
            int idx = tid + i * 256;          // 2048 float4 per tile
            int row = idx >> 4;               // 16 float4 per 64-wide row
            int c4  = idx & 15;
            const float4 v = *reinterpret_cast<const float4*>(
                x + (size_t)(m0 + row) * K + kt + c4 * 4);
            uint2 p;
            p.x = pk2(v.x, v.y);
            p.y = pk2(v.z, v.w);
            *reinterpret_cast<uint2*>(As + row * ASTR + c4 * 4) = p;
        }

        // ---- stage B: dequant int4 -> f16, store col-major ----
        {
            const int gi  = kt / gs;          // 64 | gs==128 -> constant over tile
            const int z32 = qz[(size_t)gi * NP8 + (n0 >> 3) + ng];
            const float4 s0 = *reinterpret_cast<const float4*>(sc + (size_t)gi * N + n0 + ng * 8);
            const float4 s1 = *reinterpret_cast<const float4*>(sc + (size_t)gi * N + n0 + ng * 8 + 4);
            const float sv[8] = {s0.x, s0.y, s0.z, s0.w, s1.x, s1.y, s1.z, s1.w};
            float zv[8];
            #pragma unroll
            for (int j = 0; j < 8; ++j) zv[j] = (float)((z32 >> (4 * j)) & 15);
            #pragma unroll
            for (int i = 0; i < 4; ++i) {
                const int krow = kr0 + i * 16;
                const int q32  = qw[(size_t)(kt + krow) * NP8 + (n0 >> 3) + ng];
                #pragma unroll
                for (int j = 0; j < 8; ++j) {
                    float w = ((float)((q32 >> (4 * j)) & 15) - zv[j]) * sv[j];
                    Bs[(ng * 8 + j) * BSTR + krow] = f2h(w);
                }
            }
        }

        // prefetch next k-tile (global_prefetch_b8)
        if (kt + BK < K) {
            __builtin_prefetch(x + (size_t)(m0 + (tid >> 1)) * K + kt + BK, 0, 1);
            __builtin_prefetch(qw + (size_t)(kt + BK + kr0) * NP8 + (n0 >> 3) + ng, 0, 1);
        }

        __syncthreads();

        // ---- compute: 2 k-steps of v_wmma_f32_16x16x32_f16 ----
        #pragma unroll
        for (int kk = 0; kk < BK; kk += 32) {
            v16h bfrag[2];
            #pragma unroll
            for (int nt = 0; nt < 2; ++nt) {
                // B 32x16 layout: lane -> col = lane%16, K = 8*(lane/16)+0..7 then +16..23
                const int col = wn * 32 + nt * 16 + (lane & 15);
                const int bk  = kk + (lane >> 4) * 8;
                const unsigned short* p = Bs + col * BSTR + bk;
                U16x16 t;
                t.lo = *reinterpret_cast<const u16x8*>(p);
                t.hi = *reinterpret_cast<const u16x8*>(p + 16);
                bfrag[nt] = __builtin_bit_cast(v16h, t);
            }
            #pragma unroll
            for (int mt = 0; mt < 4; ++mt) {
                // A 16x32 layout: lane -> row = lane%16, K = 8*(lane/16)+0..7 then +16..23
                const int row = wm * 64 + mt * 16 + (lane & 15);
                const int akh = kk + (lane >> 4) * 8;
                const unsigned short* p = As + row * ASTR + akh;
                U16x16 t;
                t.lo = *reinterpret_cast<const u16x8*>(p);
                t.hi = *reinterpret_cast<const u16x8*>(p + 16);
                v16h afrag = __builtin_bit_cast(v16h, t);
                #pragma unroll
                for (int nt = 0; nt < 2; ++nt) {
                    acc[mt][nt] = __builtin_amdgcn_wmma_f32_16x16x32_f16(
                        false, afrag, false, bfrag[nt],
                        (short)0, acc[mt][nt], false, false);
                }
            }
        }
        __syncthreads();
    }

    // ---- epilogue: bias add + f32 store (C/D layout: VGPR i -> M = i + 8*(lane/16), N = lane%16) ----
    #pragma unroll
    for (int mt = 0; mt < 4; ++mt) {
        #pragma unroll
        for (int nt = 0; nt < 2; ++nt) {
            const int col = n0 + wn * 32 + nt * 16 + (lane & 15);
            const float bv = bias[col];
            const int rbase = m0 + wm * 64 + mt * 16 + (lane >> 4) * 8;
            #pragma unroll
            for (int i = 0; i < 8; ++i) {
                out[(size_t)(rbase + i) * N + col] = acc[mt][nt][i] + bv;
            }
        }
    }
}

extern "C" void kernel_launch(void* const* d_in, const int* in_sizes, int n_in,
                              void* d_out, int out_size, void* d_ws, size_t ws_size,
                              hipStream_t stream) {
    const float* x    = (const float*)d_in[0];
    const int*   qw   = (const int*)d_in[1];
    const int*   qz   = (const int*)d_in[2];
    const float* sc   = (const float*)d_in[3];
    const float* bias = (const float*)d_in[4];
    float*       out  = (float*)d_out;

    // Derive shapes from flat sizes (no device reads allowed here):
    // scales [G,N], bias [N], qzeros [G,N/8], qweight [K,N/8], x [M,K]
    const int N    = in_sizes[4];
    const int G    = in_sizes[3] / N;
    const int NP8  = in_sizes[2] / G;
    const int K    = in_sizes[1] / NP8;
    const int M    = in_sizes[0] / K;
    const int gs   = K / G;

    dim3 grid(N / BN, M / BM);
    w4a16_wmma_gemm<<<grid, dim3(256), 0, stream>>>(x, qw, qz, sc, bias, out,
                                                    M, N, K, NP8, gs);
}
